// GraphSAGEProducts_19911468384535
// MI455X (gfx1250) — matrix operand order repaired and checked by reading, not compiled, not run on hardware
//
#include <hip/hip_runtime.h>

#define DIN   100
#define KP0   128
#define DH    256
#define DOUT  47
#define DOUTP 48
#define BNEPS 1e-5f

typedef __attribute__((ext_vector_type(16))) __bf16 v16bf;
typedef __attribute__((ext_vector_type(8)))  __bf16 v8bf;
typedef __attribute__((ext_vector_type(8)))  float  v8f;

static __device__ __forceinline__ __bf16 to_bf16(float f) { return (__bf16)f; }

static __device__ __forceinline__ v16bf cat16(v8bf lo, v8bf hi) {
  return __builtin_shufflevector(lo, hi, 0, 1, 2, 3, 4, 5, 6, 7, 8, 9, 10, 11, 12, 13, 14, 15);
}

// ---------------- utility kernels ----------------

__global__ void zero_f32_kernel(float* __restrict__ p, long n) {
  long i = (long)blockIdx.x * blockDim.x + threadIdx.x;
  if (i < n) p[i] = 0.0f;
}

__global__ void degree_kernel(const int* __restrict__ dst, float* __restrict__ deg, int E) {
  int e = blockIdx.x * blockDim.x + threadIdx.x;
  if (e < E) atomicAdd(&deg[dst[e]], 1.0f);
}

// scatter-add f32 features (layer 0 input), 4 features per thread
__global__ void scatter_f32_kernel(const int* __restrict__ src, const int* __restrict__ dst,
                                   const float* __restrict__ feat, float* __restrict__ agg,
                                   int E, int D, int ldA) {
  const int chunks = D >> 2;
  long t = (long)blockIdx.x * blockDim.x + threadIdx.x;
  if (t >= (long)E * chunks) return;
  int e = (int)(t / chunks);
  int c = (int)(t % chunks) << 2;
  int s = src[e], d = dst[e];
  const float* f = feat + (size_t)s * D + c;
  float* a = agg + (size_t)d * ldA + c;
  atomicAdd(a + 0, f[0]);
  atomicAdd(a + 1, f[1]);
  atomicAdd(a + 2, f[2]);
  atomicAdd(a + 3, f[3]);
}

// scatter-add bf16 features (hidden layers), 4 features per thread
__global__ void scatter_bf16_kernel(const int* __restrict__ src, const int* __restrict__ dst,
                                    const __bf16* __restrict__ feat, float* __restrict__ agg,
                                    int E, int D) {
  const int chunks = D >> 2;
  long t = (long)blockIdx.x * blockDim.x + threadIdx.x;
  if (t >= (long)E * chunks) return;
  int e = (int)(t / chunks);
  int c = (int)(t % chunks) << 2;
  int s = src[e], d = dst[e];
  const __bf16* f = feat + (size_t)s * D + c;
  float* a = agg + (size_t)d * D + c;
  atomicAdd(a + 0, (float)f[0]);
  atomicAdd(a + 1, (float)f[1]);
  atomicAdd(a + 2, (float)f[2]);
  atomicAdd(a + 3, (float)f[3]);
}

// f32 -> bf16 with zero padding (row-major activations, e.g. x: N x 100 -> N x 128)
__global__ void cvt_pad2d_bf16_kernel(const float* __restrict__ in, __bf16* __restrict__ out,
                                      int Rs, int Cs, int Rp, int Cp) {
  long i = (long)blockIdx.x * blockDim.x + threadIdx.x;
  if (i >= (long)Rp * Cp) return;
  int r = (int)(i / Cp), c = (int)(i % Cp);
  out[i] = (r < Rs && c < Cs) ? to_bf16(in[(size_t)r * Cs + c]) : to_bf16(0.0f);
}

// pack weight W[Ks x Cs] (row major f32) into WMMA B-fragment order, zero padded:
// out[((kb*CT + tn)*32 + lane)*16 + h] = bf16(W[(kb*32 + (h>>3)*16 + (lane>>4)*8 + (h&7)) * Cs + tn*16 + (lane&15)])
__global__ void pack_w_bf16_kernel(const float* __restrict__ W, __bf16* __restrict__ out,
                                   int Ks, int Cs, int Kp, int Cp) {
  long i = (long)blockIdx.x * blockDim.x + threadIdx.x;
  if (i >= (long)Kp * Cp) return;
  const int CT = Cp >> 4;
  int h    = (int)(i & 15);
  int lane = (int)((i >> 4) & 31);
  long rest = i >> 9;
  int tn = (int)(rest % CT);
  int kb = (int)(rest / CT);
  int k = kb * 32 + ((h >> 3) << 4) + ((lane >> 4) << 3) + (h & 7);
  int c = tn * 16 + (lane & 15);
  float v = (k < Ks && c < Cs) ? W[(size_t)k * Cs + c] : 0.0f;
  out[i] = to_bf16(v);
}

// mean = agg / max(deg,1), converted to bf16 (padded cols of agg are already 0)
__global__ void mean_bf16_kernel(const float* __restrict__ agg, const float* __restrict__ deg,
                                 __bf16* __restrict__ out, int n, int Dp) {
  long i = (long)blockIdx.x * blockDim.x + threadIdx.x;
  if (i >= (long)n * Dp) return;
  int r = (int)(i / Dp);
  float d = fmaxf(deg[r], 1.0f);
  out[i] = to_bf16(agg[i] / d);
}

__global__ void pad_bias_kernel(const float* __restrict__ b, float* __restrict__ bp, int n, int np) {
  int i = blockIdx.x * blockDim.x + threadIdx.x;
  if (i < np) bp[i] = (i < n) ? b[i] : 0.0f;
}

// ---------------- WMMA fused dual-GEMM: H = Amean@Wl + Aroot@Wr + bias ----------------
// One wave per 16x(16*NT) output strip; EXEC all ones as WMMA requires.
// A fragment (16x32 bf16): lane L holds row L&15; halves 0..7 = K[k0 + (L>>4)*8 .. +7],
//                          halves 8..15 = K[k0+16+(L>>4)*8 .. +7] -> two b128 loads.
// B fragments come pre-packed in fragment order: 32 contiguous bytes per lane.
// C/D: VGPR r -> M = r + 8*(lane>=16), N = lane&15.
template <int NT>
__global__ __launch_bounds__(32) void sage_wmma_gemm_kernel(
    const __bf16* __restrict__ Am, const __bf16* __restrict__ Ar,
    const __bf16* __restrict__ Wl, const __bf16* __restrict__ Wr,
    const float* __restrict__ bias, float* __restrict__ H,
    int K, int C) {
  const int tm   = blockIdx.x;
  const int tn0  = blockIdx.y * NT;
  const int lane = threadIdx.x;
  const int rid  = lane & 15;
  const int hi   = lane >> 4;
  const int CT   = C >> 4;

  const __bf16* amp = Am + (size_t)(tm * 16 + rid) * K + hi * 8;
  const __bf16* arp = Ar + (size_t)(tm * 16 + rid) * K + hi * 8;

  v8f acc[NT];
  const v8f vzero = {};
#pragma unroll
  for (int t = 0; t < NT; ++t) acc[t] = vzero;

  for (int k0 = 0; k0 < K; k0 += 32) {
    if (k0 + 32 < K) {
      __builtin_prefetch(amp + k0 + 32, 0, 1);
      __builtin_prefetch(arp + k0 + 32, 0, 1);
    }
    // A fragments: 2 x b128 each
    v16bf am = cat16(*(const v8bf*)(amp + k0), *(const v8bf*)(amp + k0 + 16));
    v16bf ar = cat16(*(const v8bf*)(arp + k0), *(const v8bf*)(arp + k0 + 16));

    const size_t wbase = ((size_t)(k0 >> 5) * CT + tn0) * 512 + (size_t)lane * 16;
    const __bf16* wlp = Wl + wbase;
    const __bf16* wrp = Wr + wbase;
#pragma unroll
    for (int t = 0; t < NT; ++t) {
      v16bf bl = *(const v16bf*)(wlp + (size_t)t * 512);
      v16bf br = *(const v16bf*)(wrp + (size_t)t * 512);
      acc[t] = __builtin_amdgcn_wmma_f32_16x16x32_bf16(false, am, false, bl, (short)0, acc[t],
                                                       false, false);
      acc[t] = __builtin_amdgcn_wmma_f32_16x16x32_bf16(false, ar, false, br, (short)0, acc[t],
                                                       false, false);
    }
  }

#pragma unroll
  for (int t = 0; t < NT; ++t) {
    const int cN = (tn0 + t) * 16 + rid;
    const float bb = bias[cN];
    float* Hp = H + (size_t)(tm * 16 + hi * 8) * C + cN;
#pragma unroll
    for (int r = 0; r < 8; ++r)
      Hp[(size_t)r * C] = acc[t][r] + bb;
  }
}

// ---------------- BatchNorm column stats (mean/var over axis 0) ----------------
__global__ __launch_bounds__(256) void col_stats_kernel(const float* __restrict__ h,
                                                        float* __restrict__ cmean,
                                                        float* __restrict__ crsig,
                                                        int n, int C) {
  __shared__ float ssum[256];
  __shared__ float ssq[256];
  const int c = blockIdx.x;
  const int t = threadIdx.x;
  float s = 0.0f, q = 0.0f;
  for (int r = t; r < n; r += 256) {
    float v = h[(size_t)r * C + c];
    s += v;
    q += v * v;
  }
  ssum[t] = s; ssq[t] = q;
  __syncthreads();
  for (int w = 128; w > 0; w >>= 1) {
    if (t < w) { ssum[t] += ssum[t + w]; ssq[t] += ssq[t + w]; }
    __syncthreads();
  }
  if (t == 0) {
    float m = ssum[0] / (float)n;
    float v = ssq[0] / (float)n - m * m;
    cmean[c] = m;
    crsig[c] = rsqrtf(v + BNEPS);
  }
}

__global__ void bn_relu_bf16_kernel(const float* __restrict__ h,
                                    const float* __restrict__ cmean, const float* __restrict__ crsig,
                                    const float* __restrict__ g, const float* __restrict__ be,
                                    __bf16* __restrict__ out, long total, int C) {
  long i = (long)blockIdx.x * blockDim.x + threadIdx.x;
  if (i >= total) return;
  int c = (int)(i % C);
  float v = (h[i] - cmean[c]) * crsig[c] * g[c] + be[c];
  out[i] = to_bf16(fmaxf(v, 0.0f));
}

// ---------------- log_softmax over 47 valid of 48 padded columns ----------------
__global__ __launch_bounds__(64) void log_softmax_kernel(const float* __restrict__ logits,
                                                         float* __restrict__ out, int n) {
  __shared__ float red[64];
  const int r = blockIdx.x;
  const int t = threadIdx.x;
  const float* L = logits + (size_t)r * DOUTP;
  float mx = -1e30f;
  for (int c = t; c < DOUT; c += 64) mx = fmaxf(mx, L[c]);
  red[t] = mx;
  __syncthreads();
  for (int w = 32; w > 0; w >>= 1) {
    if (t < w) red[t] = fmaxf(red[t], red[t + w]);
    __syncthreads();
  }
  mx = red[0];
  __syncthreads();
  float s = 0.0f;
  for (int c = t; c < DOUT; c += 64) s += expf(L[c] - mx);
  red[t] = s;
  __syncthreads();
  for (int w = 32; w > 0; w >>= 1) {
    if (t < w) red[t] += red[t + w];
    __syncthreads();
  }
  const float lse = mx + logf(red[0]);
  for (int c = t; c < DOUT; c += 64)
    out[(size_t)r * DOUT + c] = L[c] - lse;
}

// ---------------- orchestration ----------------

static inline long cdivl(long a, long b) { return (a + b - 1) / b; }

extern "C" void kernel_launch(void* const* d_in, const int* in_sizes, int n_in,
                              void* d_out, int out_size, void* d_ws, size_t ws_size,
                              hipStream_t stream) {
  const float* x   = (const float*)d_in[0];
  const int*   ei  = (const int*)d_in[1];
  const float* wl0 = (const float*)d_in[2];
  const float* wr0 = (const float*)d_in[3];
  const float* b0  = (const float*)d_in[4];
  const float* g0  = (const float*)d_in[5];
  const float* be0 = (const float*)d_in[6];
  const float* wl1 = (const float*)d_in[7];
  const float* wr1 = (const float*)d_in[8];
  const float* b1  = (const float*)d_in[9];
  const float* g1  = (const float*)d_in[10];
  const float* be1 = (const float*)d_in[11];
  const float* wl2 = (const float*)d_in[12];
  const float* wr2 = (const float*)d_in[13];
  const float* b2  = (const float*)d_in[14];

  const int N = in_sizes[0] / DIN;
  const int E = in_sizes[1] / 2;
  const int* src = ei;
  const int* dst = ei + E;

  // workspace carve-out (256B aligned blocks)
  char* w = (char*)d_ws;
  auto carve = [&](size_t bytes) {
    char* p = w;
    w += (bytes + 255) & ~(size_t)255;
    return (void*)p;
  };
  float*  deg  = (float*)carve((size_t)N * 4);
  float*  agg  = (float*)carve((size_t)N * DH * 4);   // layer0 uses first N*KP0
  float*  hf   = (float*)carve((size_t)N * DH * 4);   // pre-BN activations / logits
  __bf16* xbf  = (__bf16*)carve((size_t)N * KP0 * 2);
  __bf16* mbf  = (__bf16*)carve((size_t)N * DH * 2);  // mean features (bf16)
  __bf16* hbf  = (__bf16*)carve((size_t)N * DH * 2);  // post-BN activations (bf16)
  __bf16* wl0p = (__bf16*)carve((size_t)KP0 * DH * 2);
  __bf16* wr0p = (__bf16*)carve((size_t)KP0 * DH * 2);
  __bf16* wl1p = (__bf16*)carve((size_t)DH * DH * 2);
  __bf16* wr1p = (__bf16*)carve((size_t)DH * DH * 2);
  __bf16* wl2p = (__bf16*)carve((size_t)DH * DOUTP * 2);
  __bf16* wr2p = (__bf16*)carve((size_t)DH * DOUTP * 2);
  float*  b2p  = (float*)carve((size_t)DOUTP * 4);
  float*  cm   = (float*)carve((size_t)DH * 4);
  float*  crs  = (float*)carve((size_t)DH * 4);

  const int T = 256;
#define GRID1(total) dim3((unsigned)cdivl((long)(total), T)), dim3(T)

  // --- input conversion + weight packing (fragment order) ---
  cvt_pad2d_bf16_kernel<<<GRID1((long)N * KP0), 0, stream>>>(x, xbf, N, DIN, N, KP0);
  pack_w_bf16_kernel<<<GRID1(KP0 * DH), 0, stream>>>(wl0, wl0p, DIN, DH, KP0, DH);
  pack_w_bf16_kernel<<<GRID1(KP0 * DH), 0, stream>>>(wr0, wr0p, DIN, DH, KP0, DH);
  pack_w_bf16_kernel<<<GRID1(DH * DH), 0, stream>>>(wl1, wl1p, DH, DH, DH, DH);
  pack_w_bf16_kernel<<<GRID1(DH * DH), 0, stream>>>(wr1, wr1p, DH, DH, DH, DH);
  pack_w_bf16_kernel<<<GRID1(DH * DOUTP), 0, stream>>>(wl2, wl2p, DH, DOUT, DH, DOUTP);
  pack_w_bf16_kernel<<<GRID1(DH * DOUTP), 0, stream>>>(wr2, wr2p, DH, DOUT, DH, DOUTP);
  pad_bias_kernel<<<1, 64, 0, stream>>>(b2, b2p, DOUT, DOUTP);

  // --- degree (shared by all layers) ---
  zero_f32_kernel<<<GRID1(N), 0, stream>>>(deg, N);
  degree_kernel<<<GRID1(E), 0, stream>>>(dst, deg, E);

  // ===== Layer 0 =====
  zero_f32_kernel<<<GRID1((long)N * KP0), 0, stream>>>(agg, (long)N * KP0);
  scatter_f32_kernel<<<GRID1((long)E * (DIN / 4)), 0, stream>>>(src, dst, x, agg, E, DIN, KP0);
  mean_bf16_kernel<<<GRID1((long)N * KP0), 0, stream>>>(agg, deg, mbf, N, KP0);
  sage_wmma_gemm_kernel<4><<<dim3(N / 16, DH / 64), 32, 0, stream>>>(mbf, xbf, wl0p, wr0p, b0, hf,
                                                                     KP0, DH);
  col_stats_kernel<<<DH, 256, 0, stream>>>(hf, cm, crs, N, DH);
  bn_relu_bf16_kernel<<<GRID1((long)N * DH), 0, stream>>>(hf, cm, crs, g0, be0, hbf, (long)N * DH,
                                                          DH);

  // ===== Layer 1 =====
  zero_f32_kernel<<<GRID1((long)N * DH), 0, stream>>>(agg, (long)N * DH);
  scatter_bf16_kernel<<<GRID1((long)E * (DH / 4)), 0, stream>>>(src, dst, hbf, agg, E, DH);
  mean_bf16_kernel<<<GRID1((long)N * DH), 0, stream>>>(agg, deg, mbf, N, DH);
  sage_wmma_gemm_kernel<4><<<dim3(N / 16, DH / 64), 32, 0, stream>>>(mbf, hbf, wl1p, wr1p, b1, hf,
                                                                     DH, DH);
  col_stats_kernel<<<DH, 256, 0, stream>>>(hf, cm, crs, N, DH);
  bn_relu_bf16_kernel<<<GRID1((long)N * DH), 0, stream>>>(hf, cm, crs, g1, be1, hbf, (long)N * DH,
                                                          DH);

  // ===== Layer 2 =====
  zero_f32_kernel<<<GRID1((long)N * DH), 0, stream>>>(agg, (long)N * DH);
  scatter_bf16_kernel<<<GRID1((long)E * (DH / 4)), 0, stream>>>(src, dst, hbf, agg, E, DH);
  mean_bf16_kernel<<<GRID1((long)N * DH), 0, stream>>>(agg, deg, mbf, N, DH);
  sage_wmma_gemm_kernel<3><<<dim3(N / 16, 1), 32, 0, stream>>>(mbf, hbf, wl2p, wr2p, b2p, hf, DH,
                                                               DOUTP);
  log_softmax_kernel<<<N, 64, 0, stream>>>(hf, (float*)d_out, N);

#undef GRID1
  (void)ws_size; (void)n_in; (void)out_size;
}
